// CausalSelfAttention_58162447122842
// MI455X (gfx1250) — matrix operand order repaired and checked by reading, not compiled, not run on hardware
//
#include <hip/hip_runtime.h>
#include <math.h>

// ---------------------------------------------------------------------------
// CDNA5 (gfx1250) wave32 WMMA types
// ---------------------------------------------------------------------------
typedef __attribute__((ext_vector_type(16))) _Float16 v16h;
typedef __attribute__((ext_vector_type(8)))  _Float16 v8h;
typedef __attribute__((ext_vector_type(8)))  float    v8f;

union V16U { v16h v; v8h h[2]; };

__device__ __forceinline__ v8f wmma16(v16h a, v16h b, v8f c) {
  // D(16x16,f32) = A(16x32,f16) * B(32x16,f16) + C
  return __builtin_amdgcn_wmma_f32_16x16x32_f16(
      /*neg_a=*/false, a, /*neg_b=*/false, b,
      /*c_mod=*/(short)0, c, /*reuse_a=*/false, /*reuse_b=*/false);
}

// A-fragment (16x32 f16, M x K), ISA 7.12.2:
// lanes 0-15: M=lane, K = k0+{0..7,16..23}; lanes 16-31: M=lane-16, K = k0+{8..15,24..31}
__device__ __forceinline__ v16h load_a16(const _Float16* __restrict__ A, int ld,
                                         int row, int k0, int lane) {
  int hl = (lane >> 4) & 1;
  const _Float16* p = A + (size_t)row * ld + k0 + hl * 8;
  V16U r;
  r.h[0] = *(const v8h*)(p);
  r.h[1] = *(const v8h*)(p + 16);
  return r.v;
}

// B-fragment (32x16 f16, K x N) from a row-major [N,K] matrix (B = Wt^T):
// lanes 0-15: N=lane, K = k0+{0..7,16..23}; lanes 16-31: N=lane-16, K = k0+{8..15,24..31}
__device__ __forceinline__ v16h load_b16(const _Float16* Wt, int ld,
                                         int col, int k0, int lane) {
  int hl = (lane >> 4) & 1;
  const _Float16* p = Wt + (size_t)col * ld + k0 + hl * 8;
  V16U r;
  r.h[0] = *(const v8h*)(p);
  r.h[1] = *(const v8h*)(p + 16);
  return r.v;
}

// ---------------------------------------------------------------------------
// Problem constants
// ---------------------------------------------------------------------------
#define BB   2
#define TT   2048
#define CC   1024
#define HH   16
#define HS   64
#define MM   (BB * TT)      // 4096
#define N1   (3 * CC)       // 3072

// ---------------------------------------------------------------------------
// CDNA5 async Global->LDS path (ASYNCcnt), with synchronous fallback
// ---------------------------------------------------------------------------
#if defined(__HIP_DEVICE_COMPILE__) &&                                     \
    __has_builtin(__builtin_amdgcn_global_load_async_to_lds_b128) &&       \
    __has_builtin(__builtin_amdgcn_s_wait_asynccnt)
#define HAVE_ASYNC_LDS 1
typedef int v4i_a __attribute__((ext_vector_type(4)));
typedef __attribute__((address_space(1))) v4i_a g_v4i;   // global-AS 16B vector
typedef __attribute__((address_space(3))) v4i_a l_v4i;   // LDS-AS   16B vector
#else
#define HAVE_ASYNC_LDS 0
#endif

__device__ __forceinline__ void async_copy16B(const _Float16* g, _Float16* l) {
#if HAVE_ASYNC_LDS
  __builtin_amdgcn_global_load_async_to_lds_b128((g_v4i*)(void*)g,
                                                 (l_v4i*)(void*)l, 0, 0);
#else
  *(v8h*)l = *(const v8h*)g;   // synchronous fallback
#endif
}

template <int N>
__device__ __forceinline__ void wait_async_le() {
#if HAVE_ASYNC_LDS
  __builtin_amdgcn_s_wait_asynccnt(N);   // N is a constant expression
#endif
}

// ---------------------------------------------------------------------------
// fp32 -> f16 convert
// ---------------------------------------------------------------------------
__global__ void k_f32_to_f16(const float* __restrict__ src,
                             _Float16* __restrict__ dst, int n) {
  int i = blockIdx.x * 256 + threadIdx.x;
  if (i < n) dst[i] = (_Float16)src[i];
}

// ---------------------------------------------------------------------------
// GEMM: D[m,n] = sum_k A[m,k] * Wt[n,k]   (A:[M,K], Wt:[N,K], both row-major)
// One wave computes a 32(M) x 64(N) tile. 8 waves / block.
// ---------------------------------------------------------------------------
template <typename OutT>
__global__ __launch_bounds__(256) void k_gemm_wmma(
    const _Float16* __restrict__ A, const _Float16* __restrict__ Wt,
    OutT* __restrict__ D, int M, int N, int K) {
  int lane = threadIdx.x & 31;
  int gw = blockIdx.x * 8 + (threadIdx.x >> 5);
  int mt = M >> 5;
  int tm = gw % mt, tn = gw / mt;
  int m0 = tm * 32, n0 = tn * 64;
  if (n0 >= N) return;

  v8f acc[2][4] = {};
  int ln = lane & 15, hl = lane >> 4;
  int ra0 = m0 + ln, ra1 = m0 + 16 + ln;

  for (int k0 = 0; k0 < K; k0 += 32) {
    if (k0 + 32 < K) {  // stream-ahead hints -> global_prefetch_b8
      __builtin_prefetch(A + (size_t)ra0 * K + k0 + 32, 0, 1);
      __builtin_prefetch(Wt + (size_t)(n0 + ln) * K + k0 + 32, 0, 1);
    }
    v16h a0 = load_a16(A, K, ra0, k0, lane);
    v16h a1 = load_a16(A, K, ra1, k0, lane);
#pragma unroll
    for (int t = 0; t < 4; ++t) {
      v16h b = load_b16(Wt, K, n0 + t * 16 + ln, k0, lane);
      acc[0][t] = wmma16(a0, b, acc[0][t]);
      acc[1][t] = wmma16(a1, b, acc[1][t]);
    }
  }
  // D layout: lanes 0-15: N=lane, M=vgpr; lanes 16-31: N=lane-16, M=vgpr+8
#pragma unroll
  for (int i = 0; i < 2; ++i)
#pragma unroll
    for (int t = 0; t < 4; ++t) {
      int n = n0 + t * 16 + ln;
#pragma unroll
      for (int r = 0; r < 8; ++r) {
        int m = m0 + i * 16 + r + 8 * hl;
        D[(size_t)m * N + n] = (OutT)acc[i][t][r];
      }
    }
}

// ---------------------------------------------------------------------------
// RoPE (reference's roll-coupled variant) + pack:
//   qkv f16 [B,T,3C] -> Q,K f16 [BH,T,64] (rope'd), V^T f16 [BH,64,T]
// ---------------------------------------------------------------------------
__global__ __launch_bounds__(256) void k_rope_pack(
    const _Float16* __restrict__ qkv, _Float16* __restrict__ Q,
    _Float16* __restrict__ K, _Float16* __restrict__ Vt) {
  int idx = blockIdx.x * 256 + threadIdx.x;      // 2^21 threads
  int i = idx & 31;
  int t = (idx >> 5) & (TT - 1);
  int h = (idx >> 16) & (HH - 1);
  int b = idx >> 20;
  int bh = b * HH + h;

  size_t base = ((size_t)(b * TT + t)) * N1 + h * HS;
  const _Float16* qr = qkv + base;
  const _Float16* kr = qkv + base + CC;
  const _Float16* vr = qkv + base + 2 * CC;

  const float LN1E4_32 = 0.28782313662425574f;   // ln(10000)/32
  int i2 = (i + 1) & 31;
  int ip = (i + 31) & 31;
  float f1 = expf(-(float)i * LN1E4_32);
  float f2 = expf(-(float)i2 * LN1E4_32);
  float c1, s1, c2, s2;
  sincosf((float)t * f1, &s1, &c1);
  sincosf((float)t * f2, &s2, &c2);

  size_t ob = ((size_t)bh * TT + t) * HS;
  size_t vb0 = (size_t)bh * HS * TT + (size_t)(2 * i) * TT + t;

  // new_rot[i]  = x[2i]  *c(i) - x[2*((i-1)%32)+1]*s(i)
  // new_pass[i] = x[2i+1]*c(i) + new_rot[(i+1)%32] *s(i)
  {
    float x0 = (float)qr[2 * i], x1 = (float)qr[2 * i + 1];
    float rot_i = x0 * c1 - (float)qr[2 * ip + 1] * s1;
    float rot_n = (float)qr[2 * i2] * c2 - x1 * s2;
    Q[ob + 2 * i] = (_Float16)rot_i;
    Q[ob + 2 * i + 1] = (_Float16)(x1 * c1 + rot_n * s1);
  }
  {
    float x0 = (float)kr[2 * i], x1 = (float)kr[2 * i + 1];
    float rot_i = x0 * c1 - (float)kr[2 * ip + 1] * s1;
    float rot_n = (float)kr[2 * i2] * c2 - x1 * s2;
    K[ob + 2 * i] = (_Float16)rot_i;
    K[ob + 2 * i + 1] = (_Float16)(x1 * c1 + rot_n * s1);
  }
  Vt[vb0] = vr[2 * i];
  Vt[vb0 + TT] = vr[2 * i + 1];
}

// ---------------------------------------------------------------------------
// Fused causal flash attention, block-cooperative.
// Block = 8 waves = 8 consecutive 16-query tiles of one (b,h); the block
// double-buffers shared 32-key K/V tiles in LDS via async Global->LDS copies.
// ---------------------------------------------------------------------------

// Stage one 32-key tile: K rows [j0,j0+32) x 64 f16 (4KB) and
// V^T rows 0..63 x keys [j0,j0+32) (4KB). 256 threads x 1 chunk of 16B each.
__device__ __forceinline__ void stage_kv(const _Float16* __restrict__ Kk,
                                         const _Float16* __restrict__ Vt,
                                         _Float16* kb, _Float16* vb,
                                         int j0, int tid) {
  int krow = tid >> 3, ksub = (tid & 7) * 8;            // 32 rows x 8 chunks
  async_copy16B(Kk + (size_t)(j0 + krow) * HS + ksub, kb + krow * HS + ksub);
  int vrow = tid >> 2, vsub = (tid & 3) * 8;            // 64 rows x 4 chunks
  async_copy16B(Vt + (size_t)vrow * TT + j0 + vsub, vb + vrow * 32 + vsub);
}

__global__ __launch_bounds__(256) void k_flash_attn(
    const _Float16* __restrict__ Qg, const _Float16* __restrict__ Kg,
    const _Float16* __restrict__ Vtg, _Float16* __restrict__ Yg) {
  __shared__ __align__(32) _Float16 kbuf[2][32 * HS];   // 2 x 4 KB
  __shared__ __align__(32) _Float16 vbuf[2][HS * 32];   // 2 x 4 KB
  __shared__ __align__(32) _Float16 plds[8][16][32];    // 8 KB P bounce tiles

  int tid = threadIdx.x;
  int lane = tid & 31, wv = tid >> 5;
  int bh = blockIdx.x >> 4;                 // 32 (b,h) pairs, 16 blocks each
  int qblk = (blockIdx.x & 15) * 128;       // block covers 128 queries
  int q0 = qblk + wv * 16;                  // this wave's 16-query tile
  int ln = lane & 15, hl = lane >> 4;

  const _Float16* Q = Qg + (size_t)bh * TT * HS;
  const _Float16* Kk = Kg + (size_t)bh * TT * HS;
  const _Float16* Vt = Vtg + (size_t)bh * HS * TT;

  // Q tile A-fragments (16 x 64), reused across all key blocks
  v16h qa0 = load_a16(Q, HS, q0 + ln, 0, lane);
  v16h qa1 = load_a16(Q, HS, q0 + ln, 32, lane);

  v8f o[4] = {};
  float mrun[8], lrun[8];
#pragma unroll
  for (int r = 0; r < 8; ++r) { mrun[r] = -3.0e38f; lrun[r] = 0.f; }

  int jmax = qblk + 128;                    // causal extent of the whole block
  stage_kv(Kk, Vt, kbuf[0], vbuf[0], 0, tid);

  for (int j0 = 0; j0 < jmax; j0 += 32) {
    int buf = (j0 >> 5) & 1;
    bool more = (j0 + 32) < jmax;
    if (more) {
      stage_kv(Kk, Vt, kbuf[buf ^ 1], vbuf[buf ^ 1], j0 + 32, tid);
      wait_async_le<2>();                   // current tile (2 ops/wave) landed
    } else {
      wait_async_le<0>();
    }
    __syncthreads();                        // all waves' chunks visible

    if (j0 < q0 + 16) {                     // causal gate for this wave
      const _Float16* kb = kbuf[buf];
      const _Float16* vb = vbuf[buf];
      v8f s0 = {}, s1 = {};
      s0 = wmma16(qa0, load_b16(kb, HS, ln, 0, lane), s0);
      s0 = wmma16(qa1, load_b16(kb, HS, ln, 32, lane), s0);
      s1 = wmma16(qa0, load_b16(kb, HS, 16 + ln, 0, lane), s1);
      s1 = wmma16(qa1, load_b16(kb, HS, 16 + ln, 32, lane), s1);

      int key0 = j0 + ln, key1 = j0 + 16 + ln;
      float mnew[8], sc[8];
#pragma unroll
      for (int r = 0; r < 8; ++r) {
        int qrow = q0 + r + 8 * hl;
        float a = s0[r] * 0.125f;           // 1/sqrt(64)
        float bvl = s1[r] * 0.125f;
        if (key0 > qrow) a = -3.0e38f;      // causal mask
        if (key1 > qrow) bvl = -3.0e38f;
        float t = fmaxf(a, bvl);            // row max over 16 lanes (intra-half)
        t = fmaxf(t, __shfl_xor(t, 1, 32));
        t = fmaxf(t, __shfl_xor(t, 2, 32));
        t = fmaxf(t, __shfl_xor(t, 4, 32));
        t = fmaxf(t, __shfl_xor(t, 8, 32));
        mnew[r] = fmaxf(mrun[r], t);
        sc[r] = __expf(mrun[r] - mnew[r]);
        float p0 = __expf(a - mnew[r]);
        float p1 = __expf(bvl - mnew[r]);
        s0[r] = p0; s1[r] = p1;
        float u = p0 + p1;
        u += __shfl_xor(u, 1, 32);
        u += __shfl_xor(u, 2, 32);
        u += __shfl_xor(u, 4, 32);
        u += __shfl_xor(u, 8, 32);
        lrun[r] = lrun[r] * sc[r] + u;
        mrun[r] = mnew[r];
      }
      // rescale accumulator (stats are per-VGPR: M lives in the vgpr index)
#pragma unroll
      for (int t4 = 0; t4 < 4; ++t4)
#pragma unroll
        for (int r = 0; r < 8; ++r) o[t4][r] *= sc[r];

      // P: D-layout -> A-layout through per-wave LDS tile
#pragma unroll
      for (int r = 0; r < 8; ++r) {
        plds[wv][r + 8 * hl][ln] = (_Float16)s0[r];
        plds[wv][r + 8 * hl][ln + 16] = (_Float16)s1[r];
      }
      __builtin_amdgcn_wave_barrier();
      V16U pa;
      pa.h[0] = *(const v8h*)&plds[wv][ln][hl * 8];
      pa.h[1] = *(const v8h*)&plds[wv][ln][16 + hl * 8];
      __builtin_amdgcn_wave_barrier();

      // O += P(16x32) * V(32x64); V^T LDS tile gives contiguous B-fragments
#pragma unroll
      for (int t4 = 0; t4 < 4; ++t4) {
        v16h vbf = load_b16(vb, 32, t4 * 16 + ln, 0, lane);
        o[t4] = wmma16(pa.v, vbf, o[t4]);
      }
    }
    __syncthreads();                        // reads done before buffer reuse
  }

  // epilogue: O /= l, write y as f16 [B,T,C]
  int b = bh >> 4, h = bh & 15;
#pragma unroll
  for (int t4 = 0; t4 < 4; ++t4)
#pragma unroll
    for (int r = 0; r < 8; ++r) {
      int m = r + 8 * hl;
      float v = o[t4][r] / lrun[r];
      size_t idx = ((size_t)(b * TT + q0 + m)) * CC + h * HS + t4 * 16 + ln;
      Yg[idx] = (_Float16)v;
    }
}

// ---------------------------------------------------------------------------
// Host launcher
// ---------------------------------------------------------------------------
extern "C" void kernel_launch(void* const* d_in, const int* in_sizes, int n_in,
                              void* d_out, int out_size, void* d_ws, size_t ws_size,
                              hipStream_t stream) {
  const float* x = (const float*)d_in[0];       // [2,2048,1024]
  const float* wA = (const float*)d_in[1];      // [3072,1024]
  const float* wP = (const float*)d_in[2];      // [1024,1024]
  float* out = (float*)d_out;                   // [2,2048,1024]

  char* ws = (char*)d_ws;
  _Float16* x16 = (_Float16*)ws;   ws += (size_t)MM * CC * 2;        // 8 MB
  _Float16* wa16 = (_Float16*)ws;  ws += (size_t)N1 * CC * 2;        // 6 MB
  _Float16* wp16 = (_Float16*)ws;  ws += (size_t)CC * CC * 2;        // 2 MB
  _Float16* qkv16 = (_Float16*)ws; ws += (size_t)MM * N1 * 2;        // 24 MB
  _Float16* q16 = (_Float16*)ws;   ws += (size_t)MM * CC * 2;        // 8 MB
  _Float16* k16 = (_Float16*)ws;   ws += (size_t)MM * CC * 2;        // 8 MB
  _Float16* vt16 = (_Float16*)ws;  ws += (size_t)MM * CC * 2;        // 8 MB
  _Float16* y16 = (_Float16*)ws;   ws += (size_t)MM * CC * 2;        // 8 MB

  // 1) fp32 -> f16
  k_f32_to_f16<<<(MM * CC + 255) / 256, 256, 0, stream>>>(x, x16, MM * CC);
  k_f32_to_f16<<<(N1 * CC + 255) / 256, 256, 0, stream>>>(wA, wa16, N1 * CC);
  k_f32_to_f16<<<(CC * CC + 255) / 256, 256, 0, stream>>>(wP, wp16, CC * CC);

  // 2) qkv = x * w_attn^T   (4096 x 3072, K=1024)
  {
    int waves = (MM / 32) * (N1 / 64);          // 6144
    k_gemm_wmma<_Float16><<<waves / 8, 256, 0, stream>>>(x16, wa16, qkv16, MM, N1, CC);
  }

  // 3) RoPE + pack Q,K [BH,T,64], V^T [BH,64,T]
  {
    int threads = BB * HH * TT * 32;            // 2^21
    k_rope_pack<<<threads / 256, 256, 0, stream>>>(qkv16, q16, k16, vt16);
  }

  // 4) fused causal flash attention -> y f16 [B,T,C]
  {
    int blocks = BB * HH * (TT / 128);          // 512 (8 q-tiles per block)
    k_flash_attn<<<blocks, 256, 0, stream>>>(q16, k16, vt16, y16);
  }

  // 5) out = y * w_proj^T   (4096 x 1024, K=1024), fp32 output
  {
    int waves = (MM / 32) * (CC / 64);          // 2048
    k_gemm_wmma<float><<<waves / 8, 256, 0, stream>>>(y16, wp16, out, MM, CC, CC);
  }
}
